// Attention_23862838297095
// MI455X (gfx1250) — matrix operand order repaired and checked by reading, not compiled, not run on hardware
//
#include <hip/hip_runtime.h>
#include <hip/hip_bf16.h>

// ---------------------------------------------------------------------------
// Flash-attention (causal + padding mask) for B=8, S=2048, D=512, fp32 I/O.
// CDNA5 (gfx1250): wave32, V_WMMA_F32_16X16X32_BF16, f32 accumulation.
//
// Workgroup = 256 threads = 8 waves, handles one (batch, 32-row q-block):
//   waves 0-3 -> q-tile 0 (rows qbase..qbase+15), waves 4-7 -> q-tile 1.
//   Both tiles share each staged K/V block -> half the staging per WMMA.
// Per 32-row k-step:
//   - stage K (32 x 512 bf16, row stride 520: bank-conflict-free, 16B aligned)
//   - QK^T partials per wave (D split 4 ways, 2 score n-tiles, 8 WMMAs)
//   - restage same LDS buffer with V TRANSPOSED (d-major, k-stride 40)
//     so PV B-fragments are contiguous 32B runs -> ds_load_b128
//   - online softmax (m, l running stats), 32 threads (tile x row)
//   - P*V: 8 dense WMMAs/wave (each wave owns a 128-wide D-slice)
// ---------------------------------------------------------------------------

typedef __attribute__((ext_vector_type(16))) __bf16 v16bf;
typedef __attribute__((ext_vector_type(8)))  float  v8f;

#define B_DIM    8
#define S_LEN    2048
#define D_DIM    512
#define KSTEP    32
#define QROWS    32     // q rows per workgroup (2 tiles of 16)
#define WAVE_D   128    // D-slice per wave
#define KSTRIDE  520    // K tile row stride (elems): 1040B -> banks 4*lm, 16B aligned
#define VSTRIDE  40     // V^T k-stride (elems): 80B -> banks 4*(5*lm%16), 16B aligned
#define KVSIZE   (D_DIM * VSTRIDE)   // 20480 ushorts = 40KB (covers 32*520 too)

union Frag16 { uint4 u4[2]; v16bf bf; };

__device__ __forceinline__ unsigned short f32_to_bf16_bits(float f) {
    union { unsigned short u; __bf16 h; } cv; cv.h = (__bf16)f; return cv.u;
}
__device__ __forceinline__ unsigned int pack_bf16(float lo, float hi) {
    return (unsigned int)f32_to_bf16_bits(lo) |
           ((unsigned int)f32_to_bf16_bits(hi) << 16);
}

__global__ __launch_bounds__(256, 1)
void attention_fa_bf16_wmma(const float* __restrict__ Q,
                            const float* __restrict__ K,
                            const float* __restrict__ V,
                            const int*   __restrict__ AMASK,
                            float*       __restrict__ OUT) {
    __shared__ unsigned short KV[KVSIZE];               // 40 KB: K tile, then V^T tile
    __shared__ float          Sred[2][4][16][KSTEP];    // 16 KB  per-tile/wave QK partials
    __shared__ unsigned short Ps[2][16 * KSTEP];        // 2 KB   softmaxed probs (bf16)
    __shared__ float          Mrow[2][16];
    __shared__ float          Lrow[2][16];
    __shared__ float          Scl[2][16];

    const int tid  = threadIdx.x;        // 0..255
    const int wave = tid >> 5;           // 0..7
    const int lane = tid & 31;
    const int half = lane >> 4;
    const int lm   = lane & 15;
    const int tile = wave >> 2;          // 0 or 1: which 16-row q-tile
    const int ws   = wave & 3;           // D-slice index within tile

    const int blk    = blockIdx.x;       // 0..511
    const int b      = blk >> 6;         // batch
    const int qbase0 = (blk & 63) * QROWS;
    const int qbase  = qbase0 + tile * 16;   // this wave's q-tile base

    const float scale = 0.044194173824159216f;  // 1/sqrt(512)
    const int*  maskb = AMASK + b * S_LEN;

    // ---- Q fragments (A-matrix 16x32 bf16 layout): element j -> kk =
    //      j + 16*half; one contiguous 16-float run in global.
    v16bf Aq[4];
    {
        const float* qrow = Q + ((size_t)(b * S_LEN + qbase + lm)) * D_DIM;
        #pragma unroll
        for (int dk = 0; dk < 4; ++dk) {
            const int dbase = ws * WAVE_D + dk * 32;
            #pragma unroll
            for (int j = 0; j < 16; ++j) {
                const int kk = j + 16 * half;
                Aq[dk][j] = (__bf16)qrow[dbase + kk];
            }
        }
    }

    v8f Oacc[8];
    #pragma unroll
    for (int n = 0; n < 8; ++n)
        Oacc[n] = (v8f){0.f, 0.f, 0.f, 0.f, 0.f, 0.f, 0.f, 0.f};

    if (tid < 32) {
        Mrow[tid >> 4][tid & 15] = -__builtin_inff();
        Lrow[tid >> 4][tid & 15] = 0.0f;
    }
    __syncthreads();

    // =======================================================================
    // Both tiles need exactly the k-steps kt = 0,32,...,qbase0 (causal edge
    // handled by masking), so the loop is fully uniform across all 8 waves.
    const int kmax = qbase0 + 16;
    for (int kt = 0; kt < kmax; kt += KSTEP) {
        const size_t blkbase = ((size_t)(b * S_LEN + kt)) * D_DIM;

        // ---- Stage K block: 32 rows x 512, bf16, row stride KSTRIDE -------
        {
            const float4* kg4 = (const float4*)(K + blkbase);
            for (int idx = tid; idx < (KSTEP * D_DIM) / 4; idx += 256) {
                const int r  = idx >> 7;          // k row   (0..31)
                const int dq = idx & 127;         // d quad  (0..127)
                const float4 kf = kg4[idx];
                uint2 pk;
                pk.x = pack_bf16(kf.x, kf.y);
                pk.y = pack_bf16(kf.z, kf.w);
                *(uint2*)&KV[r * KSTRIDE + dq * 4] = pk;   // 8B aligned
            }
            if (kt + KSTEP < kmax) {   // prefetch next K/V block (L2-resident set)
                const size_t nxt = blkbase + (size_t)KSTEP * D_DIM;
                __builtin_prefetch(K + nxt + (size_t)tid * 64, 0, 1);
                __builtin_prefetch(V + nxt + (size_t)tid * 64, 0, 1);
            }
        }
        __syncthreads();

        // ---- QK^T partials: two 16x16 score tiles over wave's 128-D slice -
        // B element j -> kk = j + 16*half (contiguous 32B run per lane).
        {
            #pragma unroll
            for (int nt = 0; nt < 2; ++nt) {
                v8f acc = (v8f){0.f, 0.f, 0.f, 0.f, 0.f, 0.f, 0.f, 0.f};
                #pragma unroll
                for (int dk = 0; dk < 4; ++dk) {
                    const int dbase = ws * WAVE_D + dk * 32;
                    Frag16 Bk;
                    const uint4* p = (const uint4*)(const void*)
                        &KV[(nt * 16 + lm) * KSTRIDE + dbase + 16 * half];
                    Bk.u4[0] = p[0];
                    Bk.u4[1] = p[1];
                    acc = __builtin_amdgcn_wmma_f32_16x16x32_bf16(
                              false, Aq[dk], false, Bk.bf, (short)0, acc, false, false);
                }
                #pragma unroll
                for (int i = 0; i < 8; ++i)
                    Sred[tile][ws][i + 8 * half][nt * 16 + lm] = acc[i];
            }
        }
        __syncthreads();   // all K reads done; Sred complete

        // ---- Restage KV with V TRANSPOSED: Vt[d][r], k-stride VSTRIDE -----
        // Row pairs -> one packed b32 store per (d, r-pair).
        {
            const float4* vg4 = (const float4*)(V + blkbase);
            for (int idx = tid; idx < (KSTEP * D_DIM) / 8; idx += 256) {
                const int rp = idx >> 7;          // row pair (0..15)
                const int dq = idx & 127;         // d quad   (0..127)
                const float4 v0 = vg4[(2 * rp) * 128 + dq];
                const float4 v1 = vg4[(2 * rp + 1) * 128 + dq];
                const int d = dq * 4;
                *(unsigned int*)&KV[(d + 0) * VSTRIDE + 2 * rp] = pack_bf16(v0.x, v1.x);
                *(unsigned int*)&KV[(d + 1) * VSTRIDE + 2 * rp] = pack_bf16(v0.y, v1.y);
                *(unsigned int*)&KV[(d + 2) * VSTRIDE + 2 * rp] = pack_bf16(v0.z, v1.z);
                *(unsigned int*)&KV[(d + 3) * VSTRIDE + 2 * rp] = pack_bf16(v0.w, v1.w);
            }
        }

        // ---- Online softmax: 32 workers = (tile, q-row) -------------------
        if (tid < 32) {
            const int ts   = tid >> 4;
            const int r    = tid & 15;
            const int qrow = qbase0 + ts * 16 + r;
            float m_old = Mrow[ts][r];
            float smax  = -__builtin_inff();
            float sv[KSTEP];
            #pragma unroll
            for (int c = 0; c < KSTEP; ++c) {
                float x = (Sred[ts][0][r][c] + Sred[ts][1][r][c] +
                           Sred[ts][2][r][c] + Sred[ts][3][r][c]) * scale;
                const int kcol = kt + c;
                const bool ok = (kcol <= qrow) && (maskb[kcol] != 0);
                x = ok ? x : -__builtin_inff();
                sv[c] = x;
                smax  = fmaxf(smax, x);
            }
            const float m_new = fmaxf(m_old, smax);
            const float m_use = (m_new == -__builtin_inff()) ? 0.0f : m_new;
            const float sc    = __expf(m_old - m_use);
            float lsum = 0.0f;
            #pragma unroll
            for (int c = 0; c < KSTEP; c += 2) {
                const float p0 = __expf(sv[c]     - m_use);
                const float p1 = __expf(sv[c + 1] - m_use);
                lsum += p0 + p1;
                *(unsigned int*)&Ps[ts][r * KSTEP + c] = pack_bf16(p0, p1);
            }
            Mrow[ts][r] = m_new;
            Lrow[ts][r] = Lrow[ts][r] * sc + lsum;
            Scl[ts][r]  = sc;
        }
        __syncthreads();   // V^T staged + softmax results ready

        // ---- Rescale O accumulators by per-row factor ---------------------
        {
            float rs[8];
            #pragma unroll
            for (int i = 0; i < 8; ++i) rs[i] = Scl[tile][i + 8 * half];
            #pragma unroll
            for (int n = 0; n < 8; ++n)
                #pragma unroll
                for (int i = 0; i < 8; ++i)
                    Oacc[n][i] *= rs[i];
        }

        // ---- P * V: A = P (16x32), B = V^T slice (contiguous per lane) ----
        {
            Frag16 Ap;
            Ap.u4[0] = *(const uint4*)(const void*)&Ps[tile][lm * KSTEP + 16 * half];
            Ap.u4[1] = *(const uint4*)(const void*)&Ps[tile][lm * KSTEP + 16 * half + 8];
            #pragma unroll
            for (int n = 0; n < 8; ++n) {
                const int col = ws * WAVE_D + n * 16 + lm;
                Frag16 Bv;
                const uint4* p = (const uint4*)(const void*)
                    &KV[col * VSTRIDE + 16 * half];
                Bv.u4[0] = p[0];
                Bv.u4[1] = p[1];
                Oacc[n] = __builtin_amdgcn_wmma_f32_16x16x32_bf16(
                              false, Ap.bf, false, Bv.bf, (short)0, Oacc[n], false, false);
            }
        }
        __syncthreads();   // protect KV/Ps before next iteration's staging
    }

    // ---- Finalize: divide by row sum, write out ---------------------------
    if (tid < 32) {
        const float l = Lrow[tid >> 4][tid & 15];
        Scl[tid >> 4][tid & 15] = (l > 0.0f) ? (1.0f / l) : 0.0f;
    }
    __syncthreads();
    {
        float rs[8];
        #pragma unroll
        for (int i = 0; i < 8; ++i) rs[i] = Scl[tile][i + 8 * half];
        #pragma unroll
        for (int n = 0; n < 8; ++n) {
            const int col = ws * WAVE_D + n * 16 + lm;
            #pragma unroll
            for (int i = 0; i < 8; ++i) {
                const int row = i + 8 * half;
                OUT[((size_t)(b * S_LEN + qbase + row)) * D_DIM + col] = Oacc[n][i] * rs[i];
            }
        }
    }
}

extern "C" void kernel_launch(void* const* d_in, const int* in_sizes, int n_in,
                              void* d_out, int out_size, void* d_ws, size_t ws_size,
                              hipStream_t stream) {
    (void)in_sizes; (void)n_in; (void)out_size; (void)d_ws; (void)ws_size;
    const float* q    = (const float*)d_in[0];
    const float* k    = (const float*)d_in[1];
    const float* v    = (const float*)d_in[2];
    const int*   mask = (const int*)d_in[3];
    float*       out  = (float*)d_out;

    const int grid = B_DIM * (S_LEN / QROWS);   // 512 workgroups
    attention_fa_bf16_wmma<<<dim3(grid), dim3(256), 0, stream>>>(q, k, v, mask, out);
}
